// WordnetDGN_16286515986842
// MI455X (gfx1250) — matrix operand order repaired and checked by reading, not compiled, not run on hardware
//
#include <hip/hip_runtime.h>
#include <stdint.h>

#define NNODES 50000
#define NEDGES 800000
#define DIM    128
#define NREL   20
#define NBASE  10
#define LN_EPS 1e-12f

typedef __bf16 bf16_t;
typedef bf16_t v16bf __attribute__((ext_vector_type(16)));
typedef bf16_t v8bf  __attribute__((ext_vector_type(8)));
typedef float  v8f   __attribute__((ext_vector_type(8)));

union AFrag { v16bf v; v8bf h[2]; };

// ---- workspace layout (bytes, 128-aligned) ----
#define OFF_HB     0u          // ushort N*128            = 12,800,000
#define OFF_WBT    12800000u   // ushort R*128*128        =    655,360
#define OFF_ROOTT  13455360u   // ushort 128*128          =     32,768
#define OFF_CNT    13488128u   // int    N*R              =  4,000,000
#define OFF_SORT   17488128u   // int    E                =  3,200,000
#define OFF_RELCNT 20688128u   // int    32
#define OFF_RELOFF 20688256u   // int    32
#define OFF_CUR    20688384u   // int    32
#define OFF_TILE   20688512u   // int    32

static __device__ __forceinline__ unsigned short f32_to_bf16(float f) {
  unsigned int u = __float_as_uint(f);
  u += 0x7FFFu + ((u >> 16) & 1u);           // round-to-nearest-even
  return (unsigned short)(u >> 16);
}

// ---------------- zero ints ----------------
__global__ void k_zero(int* p, int n) {
  int t = blockIdx.x * blockDim.x + threadIdx.x;
  if (t < n) p[t] = 0;
}

// -------- W[r] = sum_b comp[r,b]*basis[b], stored transposed bf16 --------
__global__ void k_weights(const float* __restrict__ comp,
                          const float* __restrict__ basis,
                          unsigned short* __restrict__ WbT) {
  int t = blockIdx.x * blockDim.x + threadIdx.x;
  if (t >= NREL * DIM * DIM) return;
  int r = t / (DIM * DIM);
  int rem = t % (DIM * DIM);
  int i = rem / DIM;     // input dim (K)
  int o = rem % DIM;     // output dim (N)
  float acc = 0.f;
#pragma unroll
  for (int b = 0; b < NBASE; ++b)
    acc += comp[r * NBASE + b] * basis[b * DIM * DIM + i * DIM + o];
  WbT[r * DIM * DIM + o * DIM + i] = f32_to_bf16(acc);
}

// -------- rootT[o][d] = root[d][o] in bf16 --------
__global__ void k_roott(const float* __restrict__ root,
                        unsigned short* __restrict__ rootT) {
  int t = blockIdx.x * blockDim.x + threadIdx.x;
  if (t >= DIM * DIM) return;
  int d = t / DIM, o = t % DIM;
  rootT[o * DIM + d] = f32_to_bf16(root[d * DIM + o]);
}

// -------- embedding sum + LayerNorm -> bf16 h  (one wave per node) --------
__global__ void k_embed(const int* __restrict__ x,
                        const float* __restrict__ syn, const float* __restrict__ lem,
                        const float* __restrict__ pos, const float* __restrict__ sen,
                        const float* __restrict__ gamma, const float* __restrict__ beta,
                        unsigned short* __restrict__ hb) {
  int wave = threadIdx.x >> 5;
  int lane = threadIdx.x & 31;
  int n = blockIdx.x * 8 + wave;
  if (n >= NNODES) return;
  int i0 = x[n * 4 + 0];   // synset
  int i1 = x[n * 4 + 1];   // pos
  int i2 = x[n * 4 + 2];   // sense
  int i3 = x[n * 4 + 3];   // lemma
  float v[4];
  float s = 0.f;
#pragma unroll
  for (int j = 0; j < 4; ++j) {
    int e = lane + j * 32;
    v[j] = syn[(long)i0 * DIM + e] + lem[(long)i3 * DIM + e]
         + pos[(long)i1 * DIM + e] + sen[(long)i2 * DIM + e];
    s += v[j];
  }
#pragma unroll
  for (int off = 16; off > 0; off >>= 1) s += __shfl_xor(s, off, 32);
  float mu = s * (1.f / DIM);
  float q = 0.f;
#pragma unroll
  for (int j = 0; j < 4; ++j) { float d = v[j] - mu; q += d * d; }
#pragma unroll
  for (int off = 16; off > 0; off >>= 1) q += __shfl_xor(q, off, 32);
  float rs = rsqrtf(q * (1.f / DIM) + LN_EPS);
#pragma unroll
  for (int j = 0; j < 4; ++j) {
    int e = lane + j * 32;
    float hv = (v[j] - mu) * rs * gamma[e] + beta[e];
    hb[(long)n * DIM + e] = f32_to_bf16(hv);
  }
}

// -------- per-(dst,rel) counts + per-relation counts --------
__global__ void k_count(const int* __restrict__ EI, const int* __restrict__ ea,
                        int* __restrict__ cnt, int* __restrict__ relCnt) {
  int e = blockIdx.x * blockDim.x + threadIdx.x;
  if (e >= NEDGES) return;
  int rel = ea[e];
  int dst = EI[NEDGES + e];
  atomicAdd(&cnt[dst * NREL + rel], 1);
  atomicAdd(&relCnt[rel], 1);
}

// -------- tiny scan: bucket offsets + tile offsets --------
__global__ void k_scan(const int* __restrict__ relCnt,
                       int* __restrict__ relOff, int* __restrict__ tileOff) {
  if (threadIdx.x != 0 || blockIdx.x != 0) return;
  relOff[0] = 0; tileOff[0] = 0;
  for (int r = 0; r < NREL; ++r) {
    relOff[r + 1] = relOff[r] + relCnt[r];
    tileOff[r + 1] = tileOff[r] + (relCnt[r] + 15) / 16;
  }
}

// -------- scatter edge ids into relation buckets --------
__global__ void k_scatter(const int* __restrict__ ea,
                          const int* __restrict__ relOff,
                          int* __restrict__ relCursor, int* __restrict__ sorted) {
  int e = blockIdx.x * blockDim.x + threadIdx.x;
  if (e >= NEDGES) return;
  int r = ea[e];
  int p = atomicAdd(&relCursor[r], 1);
  sorted[relOff[r] + p] = e;
}

// -------- out = h @ root + bias   (WMMA bf16, one wave = 16 rows) --------
__global__ void k_root_gemm(const unsigned short* __restrict__ hb,
                            const unsigned short* __restrict__ rootT,
                            const float* __restrict__ bias,
                            float* __restrict__ out) {
  int lane = threadIdx.x & 31;
  int hi = lane >> 4;        // K-half selector
  int col = lane & 15;       // A-row / B-col
  int base = blockIdx.x * 16;

  v8f acc[8];
#pragma unroll
  for (int nt = 0; nt < 8; ++nt) acc[nt] = (v8f){0.f,0.f,0.f,0.f,0.f,0.f,0.f,0.f};

#pragma unroll
  for (int kc = 0; kc < 4; ++kc) {
    AFrag a;
    const unsigned short* ap = hb + (long)(base + col) * DIM + kc * 32 + hi * 16;
    a.h[0] = *(const v8bf*)(ap);
    a.h[1] = *(const v8bf*)(ap + 8);
#pragma unroll
    for (int nt = 0; nt < 8; ++nt) {
      const unsigned short* bp = rootT + (nt * 16 + col) * DIM + kc * 32 + hi * 16;
      v16bf b = *(const v16bf*)bp;
      acc[nt] = __builtin_amdgcn_wmma_f32_16x16x32_bf16(
          false, a.v, false, b, (short)0, acc[nt], false, false);
    }
  }
#pragma unroll
  for (int g = 0; g < 8; ++g) {
    int m = g + 8 * hi;
    long row = base + m;
#pragma unroll
    for (int nt = 0; nt < 8; ++nt)
      out[row * DIM + nt * 16 + col] = acc[nt][g] + bias[nt * 16 + col];
  }
}

// -------- edge tiles: 16 same-relation edges -> 16x128 GEMM + scaled atomics --------
__global__ void k_edge_gemm(const unsigned short* __restrict__ hb,
                            const unsigned short* __restrict__ WbT,
                            const int* __restrict__ EI,
                            const int* __restrict__ sorted,
                            const int* __restrict__ cnt,
                            const int* __restrict__ relOff,
                            const int* __restrict__ tileOff,
                            float* __restrict__ out) {
  int t = blockIdx.x;
  int r = -1, lt = 0;
#pragma unroll
  for (int i = 0; i < NREL; ++i)
    if (t >= tileOff[i] && t < tileOff[i + 1]) { r = i; lt = t - tileOff[i]; }
  if (r < 0) return;                                   // wave-uniform exit

  int estart = relOff[r] + lt * 16;
  int valid = relOff[r + 1] - estart;
  if (valid > 16) valid = 16;

  int lane = threadIdx.x & 31;
  int hi = lane >> 4;
  int col = lane & 15;       // this lane's edge slot (and B column)

  int e = sorted[estart + (col < valid ? col : 0)];
  int src = EI[e];
  int dst = EI[NEDGES + e];
  int c = cnt[dst * NREL + r];
  float inv = 1.0f / (float)(c > 0 ? c : 1);

  v8f acc[8];
#pragma unroll
  for (int nt = 0; nt < 8; ++nt) acc[nt] = (v8f){0.f,0.f,0.f,0.f,0.f,0.f,0.f,0.f};

  const unsigned short* wbase = WbT + (long)r * DIM * DIM;
#pragma unroll
  for (int kc = 0; kc < 4; ++kc) {
    AFrag a;
    const unsigned short* ap = hb + (long)src * DIM + kc * 32 + hi * 16;
    a.h[0] = *(const v8bf*)(ap);
    a.h[1] = *(const v8bf*)(ap + 8);
#pragma unroll
    for (int nt = 0; nt < 8; ++nt) {
      const unsigned short* bp = wbase + (nt * 16 + col) * DIM + kc * 32 + hi * 16;
      v16bf b = *(const v16bf*)bp;
      acc[nt] = __builtin_amdgcn_wmma_f32_16x16x32_bf16(
          false, a.v, false, b, (short)0, acc[nt], false, false);
    }
  }

  // scaled atomic accumulation: element (m, nt*16+col) in lane, VGPR g
#pragma unroll
  for (int g = 0; g < 8; ++g) {
    int m = g + 8 * hi;
    int dstm = __shfl(dst, m, 32);
    float invm = __shfl(inv, m, 32);
    bool ok = (m < valid);
#pragma unroll
    for (int nt = 0; nt < 8; ++nt) {
      if (ok) atomicAdd(&out[(long)dstm * DIM + nt * 16 + col], acc[nt][g] * invm);
    }
  }
}

extern "C" void kernel_launch(void* const* d_in, const int* in_sizes, int n_in,
                              void* d_out, int out_size, void* d_ws, size_t ws_size,
                              hipStream_t stream) {
  const int*   x     = (const int*)d_in[0];
  const int*   EI    = (const int*)d_in[1];
  const int*   ea    = (const int*)d_in[2];
  const float* syn   = (const float*)d_in[3];
  const float* lem   = (const float*)d_in[4];
  const float* pos   = (const float*)d_in[5];
  const float* sen   = (const float*)d_in[6];
  const float* gam   = (const float*)d_in[7];
  const float* bet   = (const float*)d_in[8];
  const float* basis = (const float*)d_in[9];
  const float* comp  = (const float*)d_in[10];
  const float* root  = (const float*)d_in[11];
  const float* bias  = (const float*)d_in[12];
  float* out = (float*)d_out;

  char* w = (char*)d_ws;
  unsigned short* hb    = (unsigned short*)(w + OFF_HB);
  unsigned short* WbT   = (unsigned short*)(w + OFF_WBT);
  unsigned short* rootT = (unsigned short*)(w + OFF_ROOTT);
  int* cnt       = (int*)(w + OFF_CNT);
  int* sorted    = (int*)(w + OFF_SORT);
  int* relCnt    = (int*)(w + OFF_RELCNT);
  int* relOff    = (int*)(w + OFF_RELOFF);
  int* relCursor = (int*)(w + OFF_CUR);
  int* tileOff   = (int*)(w + OFF_TILE);

  // zero counters (cnt[N*R], relCnt/relCursor)
  k_zero<<<(NNODES * NREL + 255) / 256, 256, 0, stream>>>(cnt, NNODES * NREL);
  k_zero<<<1, 128, 0, stream>>>(relCnt, 128);   // covers relCnt/relOff/relCursor/tileOff region

  // weights -> bf16 transposed
  k_weights<<<(NREL * DIM * DIM + 255) / 256, 256, 0, stream>>>(comp, basis, WbT);
  k_roott<<<(DIM * DIM + 255) / 256, 256, 0, stream>>>(root, rootT);

  // node features: embed + LayerNorm -> bf16
  k_embed<<<(NNODES + 7) / 8, 256, 0, stream>>>(x, syn, lem, pos, sen, gam, bet, hb);

  // segment counts + relation histogram
  k_count<<<(NEDGES + 255) / 256, 256, 0, stream>>>(EI, ea, cnt, relCnt);
  k_scan<<<1, 1, 0, stream>>>(relCnt, relOff, tileOff);
  k_scatter<<<(NEDGES + 255) / 256, 256, 0, stream>>>(ea, relOff, relCursor, sorted);

  // out = h @ root + bias  (initializes d_out)
  k_root_gemm<<<NNODES / 16, 32, 0, stream>>>(hb, rootT, bias, out);

  // + per-relation mean-aggregated messages (atomic)
  k_edge_gemm<<<NEDGES / 16 + NREL, 32, 0, stream>>>(hb, WbT, EI, sorted, cnt,
                                                     relOff, tileOff, out);
}
